// memristor_dense_14998025798447
// MI455X (gfx1250) — matrix operand order repaired and checked by reading, not compiled, not run on hardware
//
#include <hip/hip_runtime.h>

// Memristor-dense layer recast as f16 WMMA GEMM for gfx1250 (MI455X).
//
// y[b,m] = (1/(K_V*k_G)) * sum_{l=1..3} sum_n C_l[n,m] * I_at(l,b,n)
//   A  (128 x 3168 f16): per-level interpolated currents of inputs (scale folded)
//   CT (1024 x 3168 f16): per-weight level coefficients, stored K-contiguous
// Contraction K = 3 levels * 1056 (1025 padded to mult. of 32).
//
// ws layout (bytes):
//   [0,     2048) : 512 partial abs-max floats
//   [2048,  2056) : params {k_G, 1/(K_V*k_G)}
//   [8192,  +811008)   : A  f16  (128*3168)
//   [819200,+6488064)  : CT f16  (1024*3168)   -> total ~7.31 MB

typedef __attribute__((ext_vector_type(16))) _Float16 v16h;
typedef __attribute__((ext_vector_type(8)))  _Float16 v8h;
typedef __attribute__((ext_vector_type(8)))  float    v8f;

#define NIN    1024
#define NOUT   1024
#define BATCH  128
#define N1     1025           // NIN + ones/bias row
#define NPAD   1056           // N1 padded to multiple of 32
#define KTOT   (3 * NPAD)     // 3168
#define G_MIN_C 0.0001f
#define G_MAX_C 0.0005f
#define K_V_C   1.05f

__constant__ float c_IREF[3][9] = {
  {-0.00015f, -0.00011f, -7e-05f, -3e-05f, 0.f, 3e-05f, 7e-05f, 0.00011f, 0.00015f},
  {-0.0005f,  -0.00035f, -0.00022f, -9e-05f, 0.f, 9e-05f, 0.00022f, 0.00035f, 0.0005f},
  {-0.0009f,  -0.0006f,  -0.00037f, -0.00015f, 0.f, 0.00015f, 0.00037f, 0.0006f, 0.0009f}
};

// ---- 1) abs-max over {w, b}: partial per block -------------------------------
__global__ void k_absmax_partial(const float* __restrict__ w,
                                 const float* __restrict__ b,
                                 float* __restrict__ partial) {
  __shared__ float sm[256];
  const int total = NIN * NOUT + NOUT;
  float m = 0.f;
  for (int i = blockIdx.x * 256 + threadIdx.x; i < total; i += 256 * 512) {
    float v = (i < NIN * NOUT) ? w[i] : b[i - NIN * NOUT];
    m = fmaxf(m, fabsf(v));
  }
  sm[threadIdx.x] = m;
  __syncthreads();
  for (int s = 128; s > 0; s >>= 1) {
    if (threadIdx.x < s) sm[threadIdx.x] = fmaxf(sm[threadIdx.x], sm[threadIdx.x + s]);
    __syncthreads();
  }
  if (threadIdx.x == 0) partial[blockIdx.x] = sm[0];
}

// ---- 2) finalize reduction, write params ------------------------------------
__global__ void k_absmax_final(const float* __restrict__ partial,
                               float* __restrict__ params) {
  __shared__ float sm[512];
  sm[threadIdx.x] = partial[threadIdx.x];
  __syncthreads();
  for (int s = 256; s > 0; s >>= 1) {
    if (threadIdx.x < s) sm[threadIdx.x] = fmaxf(sm[threadIdx.x], sm[threadIdx.x + s]);
    __syncthreads();
  }
  if (threadIdx.x == 0) {
    float mw = sm[0];
    float kG = (G_MAX_C - G_MIN_C) / mw;
    params[0] = kG;
    params[1] = 1.f / (K_V_C * kG);
  }
}

// ---- 3) build A: interpolated currents per level, scale folded ---------------
__global__ void k_build_A(const float* __restrict__ x,
                          const float* __restrict__ params,
                          _Float16* __restrict__ A) {
  int tid = blockIdx.x * blockDim.x + threadIdx.x;
  if (tid >= BATCH * NPAD) return;
  const int n  = tid % NPAD;
  const int bi = tid / NPAD;
  float out0 = 0.f, out1 = 0.f, out2 = 0.f;
  if (n < N1) {
    const float inv_scale = params[1];
    const float xv = (n < NIN) ? x[bi * NIN + n] : 1.0f;
    const float V = K_V_C * xv;
    float u = (V + 1.2f) * (1.0f / 0.3f);
    int i = (int)floorf(u);
    i = max(0, min(7, i));
    float f = fminf(fmaxf(u - (float)i, 0.f), 1.f);
    out0 = (c_IREF[0][i] + f * (c_IREF[0][i + 1] - c_IREF[0][i])) * inv_scale;
    out1 = (c_IREF[1][i] + f * (c_IREF[1][i + 1] - c_IREF[1][i])) * inv_scale;
    out2 = (c_IREF[2][i] + f * (c_IREF[2][i + 1] - c_IREF[2][i])) * inv_scale;
  }
  _Float16* row = A + bi * KTOT + n;
  row[0 * NPAD] = (_Float16)out0;
  row[1 * NPAD] = (_Float16)out1;
  row[2 * NPAD] = (_Float16)out2;
}

// ---- 4) build CT: per-weight level coefficients, K-contiguous layout ---------
__global__ void k_build_CT(const float* __restrict__ w,
                           const float* __restrict__ b,
                           const float* __restrict__ params,
                           _Float16* __restrict__ CT) {
  int tid = blockIdx.x * blockDim.x + threadIdx.x;
  if (tid >= NOUT * NPAD) return;
  const int n = tid % NPAD;
  const int m = tid / NPAD;
  float c0 = 0.f, c1 = 0.f, c2 = 0.f;
  if (n < N1) {
    const float kG = params[0];
    const float wv = (n < NIN) ? w[n * NOUT + m] : b[m];
    const float G  = kG * wv;
    const float Ga = fabsf(G);
    const float sgn = (G > 0.f) ? 1.f : ((G < 0.f) ? -1.f : 0.f);
    // searchsorted(G_REF, Ga, 'right') - 1, clipped to [0,2]
    int idx = (Ga >= 3e-4f) ? 2 : ((Ga >= 1e-4f) ? 1 : 0);
    const float g0 = (idx == 0) ? 0.f    : ((idx == 1) ? 1e-4f : 3e-4f);
    const float g1 = (idx == 0) ? 1e-4f  : ((idx == 1) ? 3e-4f : 5e-4f);
    const float t = (Ga - g0) / (g1 - g0);
    // level idx   gets sgn*(1-t)  (level 0 current == 0, so drop when idx==0)
    // level idx+1 gets sgn*t      (slot j = level-1 in 0..2)
    const float lo = sgn * (1.f - t);
    const float hi = sgn * t;
    if (idx == 0)      {            c0 = hi; }
    else if (idx == 1) { c0 = lo;   c1 = hi; }
    else               { c1 = lo;   c2 = hi; }
  }
  _Float16* row = CT + m * KTOT + n;
  row[0 * NPAD] = (_Float16)c0;
  row[1 * NPAD] = (_Float16)c1;
  row[2 * NPAD] = (_Float16)c2;
}

// ---- 5) WMMA GEMM: y(128x1024 f32) = A(128xK f16) * CT^T(Kx1024 f16) ---------
// One wave per 16x16 output tile; 99 K-steps of v_wmma_f32_16x16x32_f16.
// No divergence / early-exit: EXEC stays all-ones (WMMA requirement).
__global__ void __launch_bounds__(32)
k_wmma_gemm(const _Float16* __restrict__ A,
            const _Float16* __restrict__ CT,
            float* __restrict__ y) {
  const int lane  = threadIdx.x;        // 0..31
  const int tileN = blockIdx.x;         // 0..63
  const int tileM = blockIdx.y;         // 0..7
  const int lm    = lane & 15;
  const int grp   = lane >> 4;          // 0 or 1
  const int hiA   = grp * 16;           // A fragment K offset for upper lanes
  const int loB   = grp * 16;           // B fragment K offset for upper lanes

  const _Float16* arow = A  + (tileM * 16 + lm) * KTOT;
  const _Float16* brow = CT + (tileN * 16 + lm) * KTOT;

  v8f acc = {};
  for (int kb = 0; kb < KTOT; kb += 32) {
    // A 16x32 f16 fragment: lane holds K = {hiA..hiA+15}
    v8h a0 = *(const v8h*)(arow + kb + hiA);
    v8h a1 = *(const v8h*)(arow + kb + hiA + 8);
    // B 32x16 f16 fragment (from CT, K-contiguous): K = {loB..loB+15}
    v16h bv = *(const v16h*)(brow + kb + loB);
    v16h av;
#pragma unroll
    for (int i = 0; i < 8; ++i) { av[i] = a0[i]; av[8 + i] = a1[i]; }
    acc = __builtin_amdgcn_wmma_f32_16x16x32_f16(
        /*neg_a=*/false, av, /*neg_b=*/false, bv,
        /*c_mod=*/(short)0, acc, /*reuse_a=*/false, /*reuse_b=*/false);
  }

  // D layout: VGPR r -> M = r + 8*grp, N = lane&15
  const int row0 = tileM * 16 + grp * 8;
  const int col  = tileN * 16 + lm;
#pragma unroll
  for (int r = 0; r < 8; ++r) y[(row0 + r) * NOUT + col] = acc[r];
}

extern "C" void kernel_launch(void* const* d_in, const int* in_sizes, int n_in,
                              void* d_out, int out_size, void* d_ws, size_t ws_size,
                              hipStream_t stream) {
  const float* x = (const float*)d_in[0];   // (128, 1024)
  const float* w = (const float*)d_in[1];   // (1024, 1024)
  const float* b = (const float*)d_in[2];   // (1024,)
  float* y = (float*)d_out;                 // (128, 1024) f32

  char* ws = (char*)d_ws;                   // needs ~7.31 MB
  float*    partial = (float*)ws;           // 512 floats
  float*    params  = (float*)(ws + 2048);  // {k_G, 1/(K_V*k_G)}
  _Float16* A       = (_Float16*)(ws + 8192);
  _Float16* CT      = (_Float16*)(ws + 819200);

  k_absmax_partial<<<512, 256, 0, stream>>>(w, b, partial);
  k_absmax_final  <<<1,   512, 0, stream>>>(partial, params);
  k_build_A <<<(BATCH * NPAD + 255) / 256, 256, 0, stream>>>(x, params, A);
  k_build_CT<<<(NOUT  * NPAD + 255) / 256, 256, 0, stream>>>(w, b, params, CT);
  k_wmma_gemm<<<dim3(64, 8), 32, 0, stream>>>(A, CT, y);
}
